// Bonv_56994216017978
// MI455X (gfx1250) — matrix operand order repaired and checked by reading, not compiled, not run on hardware
//
#include <hip/hip_runtime.h>
#include <cstdint>
#include <cstddef>

#define NN 8192
#define CD 128

typedef __attribute__((ext_vector_type(16))) __bf16        v16bf;
typedef __attribute__((ext_vector_type(8)))  float         v8f;
typedef __attribute__((ext_vector_type(8)))  unsigned int  v8u;

__device__ __forceinline__ unsigned short f2bf_u(float f) {
  unsigned u = __builtin_bit_cast(unsigned, f);
  u += 0x7FFFu + ((u >> 16) & 1u);            // round-to-nearest-even
  return (unsigned short)(u >> 16);
}
__device__ __forceinline__ unsigned pack_bf2(float lo, float hi) {
  return (unsigned)f2bf_u(lo) | ((unsigned)f2bf_u(hi) << 16);
}

// ---------------------------------------------------------------------------
// Kernel 1: column reduction of A (deg + masked feature sums), per-row-chunk
// partials (no atomics -> deterministic). 1024 blocks x 256 threads.
// block = (chunk of 256 rows) x (256 columns). Coalesced across columns.
// ---------------------------------------------------------------------------
__global__ void k_colred_partial(const float* __restrict__ A,
                                 const float* __restrict__ nodes,
                                 float* __restrict__ pd,
                                 float* __restrict__ pa0,
                                 float* __restrict__ pa1) {
  const int colblk = blockIdx.x & 31;
  const int chunk  = blockIdx.x >> 5;
  const int col    = colblk * 256 + threadIdx.x;
  const int j0     = chunk * 256;
  float d = 0.f, a0 = 0.f, a1 = 0.f;
  for (int jj = 0; jj < 256; ++jj) {
    const int j = j0 + jj;
    const float v = A[(size_t)j * NN + col];
    if (v != 0.f) {
      d  += 1.f;
      a0 += nodes[2 * j];
      a1 += nodes[2 * j + 1];
    }
  }
  pd [chunk * NN + col] = d;
  pa0[chunk * NN + col] = a0;
  pa1[chunk * NN + col] = a1;
}

// Combine the 32 chunk partials in fixed order. 32 blocks x 256.
__global__ void k_colred_final(const float* __restrict__ pd,
                               const float* __restrict__ pa0,
                               const float* __restrict__ pa1,
                               float* __restrict__ deg,
                               float* __restrict__ agg) {
  const int col = blockIdx.x * 256 + threadIdx.x;
  float d = 0.f, a0 = 0.f, a1 = 0.f;
  for (int ch = 0; ch < 32; ++ch) {
    d  += pd [ch * NN + col];
    a0 += pa0[ch * NN + col];
    a1 += pa1[ch * NN + col];
  }
  deg[col]         = d;
  agg[2 * col]     = a0;
  agg[2 * col + 1] = a1;
}

// ---------------------------------------------------------------------------
// Kernel 2: fused sage1 + sage2 per node. 32 blocks x 256 (one node/thread).
// sage: out = (agg/deg) @ Wl.T + bl + x @ Wr.T ; row L2-normalize.
// x branch -> x_feat (f32). s branch -> softmax -> s_soft (f32) + s_bf
// (bf16, GEMM-tiled layout) + per-node entropy term.
//
// s_bf tiled layout (ushort): [K-tile kt = n>>5][col c][pair (n>>1)&15][n&1]
// -> each 32-K GEMM tile is one contiguous 8 KB block, and each wave (32
// consecutive nodes) writes one contiguous 64 B chunk per column: coalesced.
// ---------------------------------------------------------------------------
__global__ void k_sage12(const float* __restrict__ nodes,
                         const float* __restrict__ deg,
                         const float* __restrict__ agg,
                         const float* __restrict__ Wl1, const float* __restrict__ bl1,
                         const float* __restrict__ Wr1,
                         const float* __restrict__ Wl2, const float* __restrict__ bl2,
                         const float* __restrict__ Wr2,
                         float* __restrict__ x_feat,
                         float* __restrict__ s_soft,
                         unsigned short* __restrict__ s_bf,
                         float* __restrict__ ent_node) {
  const int n = blockIdx.x * 256 + threadIdx.x;
  const float d   = deg[n];
  const float inv = d > 0.f ? 1.f / fmaxf(d, 1.f) : 0.f;
  const float a0 = agg[2 * n] * inv, a1 = agg[2 * n + 1] * inv;
  const float x0 = nodes[2 * n],     x1 = nodes[2 * n + 1];

  // ---- x branch (Wl1/Wr1): recompute pattern keeps VGPR use low ----
  float ss = 0.f;
  for (int k = 0; k < CD; ++k) {
    const float o = a0 * Wl1[2 * k] + a1 * Wl1[2 * k + 1] + bl1[k]
                  + x0 * Wr1[2 * k] + x1 * Wr1[2 * k + 1];
    ss += o * o;
  }
  const float rn = 1.f / fmaxf(sqrtf(ss), 1e-12f);
  for (int k = 0; k < CD; ++k) {
    const float o = a0 * Wl1[2 * k] + a1 * Wl1[2 * k + 1] + bl1[k]
                  + x0 * Wr1[2 * k] + x1 * Wr1[2 * k + 1];
    x_feat[(size_t)n * CD + k] = o * rn;
  }

  // ---- s branch (Wl2/Wr2): normalize then stable softmax ----
  float ss2 = 0.f;
  for (int k = 0; k < CD; ++k) {
    const float o = a0 * Wl2[2 * k] + a1 * Wl2[2 * k + 1] + bl2[k]
                  + x0 * Wr2[2 * k] + x1 * Wr2[2 * k + 1];
    ss2 += o * o;
  }
  const float rn2 = 1.f / fmaxf(sqrtf(ss2), 1e-12f);
  float mx = -3.4e38f;
  for (int k = 0; k < CD; ++k) {
    const float o = a0 * Wl2[2 * k] + a1 * Wl2[2 * k + 1] + bl2[k]
                  + x0 * Wr2[2 * k] + x1 * Wr2[2 * k + 1];
    mx = fmaxf(mx, o * rn2);
  }
  float es = 0.f;
  for (int k = 0; k < CD; ++k) {
    const float o = a0 * Wl2[2 * k] + a1 * Wl2[2 * k + 1] + bl2[k]
                  + x0 * Wr2[2 * k] + x1 * Wr2[2 * k + 1];
    es += expf(o * rn2 - mx);
  }
  const float ies = 1.f / es;
  // tiled bf16 destination base for this node
  const size_t sb_base = (size_t)(n >> 5) * 4096 + ((n >> 1) & 15) * 2 + (n & 1);
  float ent = 0.f;
  for (int k = 0; k < CD; ++k) {
    const float o = a0 * Wl2[2 * k] + a1 * Wl2[2 * k + 1] + bl2[k]
                  + x0 * Wr2[2 * k] + x1 * Wr2[2 * k + 1];
    const float sv = expf(o * rn2 - mx) * ies;
    s_soft[(size_t)n * CD + k] = sv;
    s_bf  [sb_base + (size_t)k * 32] = f2bf_u(sv);
    ent -= sv * logf(sv + 1e-15f);
  }
  ent_node[n] = ent;
}

// ---------------------------------------------------------------------------
// Kernel 3: P = A(8192x8192, f32 0/1) @ s(8192x128, bf16) via WMMA bf16.
// 512 blocks x 256 threads (8 waves). Block owns 16 rows x 128 cols.
// Each wave: one 16x16 C tile, K-loop of 256 x v_wmma_f32_16x16x32_bf16.
// Double-buffered LDS; A converted to packed bf16x2 ONCE at staging (not
// per wave); A and B fragments are each exactly 2 x ds_load_b128 per lane.
// Fragment layouts per ISA 7.12.2:
//   A 16x32 bf16: lane -> row M=lo; vreg i -> K pair (i<4 ? i : i+4) + 4*hi
//   B 32x16 bf16: lane -> col N=lo; vreg i -> K pair (i<4 ? i : i+4) + 4*hi
//   C 16x16 f32 : lane -> N=lo; vreg r -> M = r + 8*hi
// ---------------------------------------------------------------------------
__global__ void k_gemm_bf16(const float* __restrict__ A,
                            const unsigned short* __restrict__ s_bf,
                            float* __restrict__ P) {
  __shared__ unsigned lds_a[2][16 * 16];   // [row][pair] bf16x2, 1 KB/buf
  __shared__ unsigned lds_s[2][128 * 16];  // [col][pair] bf16x2, 8 KB/buf

  const int tid  = threadIdx.x;
  const int m0   = blockIdx.x * 16;
  const int n0   = (tid >> 5) * 16;        // wave id * 16
  const int lane = tid & 31;
  const int hi   = lane >> 4;
  const int lo   = lane & 15;

  auto load_tiles = [&](int kt, int buf) {
    const int k0 = kt * 32;
    // A tile: 16 rows x 16 pairs. One float2 load + pack per thread.
    const int r = tid >> 4;                // row 0..15
    const int t = tid & 15;                // pair 0..15
    const float2 av = *reinterpret_cast<const float2*>(
        A + (size_t)(m0 + r) * NN + k0 + 2 * t);
    lds_a[buf][r * 16 + t] = pack_bf2(av.x, av.y);
    // s tile: contiguous 8 KB block (tiled global layout), b128 copies.
    const uint4* sp = reinterpret_cast<const uint4*>(s_bf + (size_t)kt * 4096);
    uint4* dp = reinterpret_cast<uint4*>(&lds_s[buf][0]);
    dp[tid]       = sp[tid];
    dp[tid + 256] = sp[tid + 256];
  };

  v8f acc = {};
  load_tiles(0, 0);
  __syncthreads();

  for (int kt = 0; kt < NN / 32; ++kt) {
    const int cur = kt & 1;
    if (kt + 1 < NN / 32) load_tiles(kt + 1, cur ^ 1);   // prefetch next

    const uint4* a4 = reinterpret_cast<const uint4*>(&lds_a[cur][0]);
    const uint4* b4 = reinterpret_cast<const uint4*>(&lds_s[cur][0]);
    const uint4 aL = a4[lo * 4 + hi];                 // pairs 0..3  (+4*hi)
    const uint4 aH = a4[lo * 4 + 2 + hi];             // pairs 8..11 (+4*hi)
    const uint4 bL = b4[(n0 + lo) * 4 + hi];          // pairs 0..3  (+4*hi)
    const uint4 bH = b4[(n0 + lo) * 4 + 2 + hi];      // pairs 8..11 (+4*hi)

    v8u au, bu;
    au[0] = aL.x; au[1] = aL.y; au[2] = aL.z; au[3] = aL.w;
    au[4] = aH.x; au[5] = aH.y; au[6] = aH.z; au[7] = aH.w;
    bu[0] = bL.x; bu[1] = bL.y; bu[2] = bL.z; bu[3] = bL.w;
    bu[4] = bH.x; bu[5] = bH.y; bu[6] = bH.z; bu[7] = bH.w;
    const v16bf a = __builtin_bit_cast(v16bf, au);
    const v16bf b = __builtin_bit_cast(v16bf, bu);

    acc = __builtin_amdgcn_wmma_f32_16x16x32_bf16(
        false, a, false, b, (short)0, acc, false, false);
    __syncthreads();
  }

#pragma unroll
  for (int r = 0; r < 8; ++r)
    P[(size_t)(m0 + r + 8 * hi) * CD + n0 + lo] = acc[r];
}

// ---------------------------------------------------------------------------
// Kernel 4: C[c][f] = sum_n A[n][c]*B[n][f]  (C = A^T B, A,B are [K,128]).
// 64 blocks (8x8 tiles of 16x16) x 256 threads. Optionally emits per-block
// sum-of-squares partials (Frobenius of C) with no atomics.
// ---------------------------------------------------------------------------
__global__ void k_gemm_tn(const float* __restrict__ Am,
                          const float* __restrict__ Bm,
                          float* __restrict__ C,
                          float* __restrict__ fro_part,
                          int K) {
  __shared__ float red[256];
  const int tid = threadIdx.x;
  const int tx = tid & 15, ty = tid >> 4;
  const int c = ((blockIdx.x >> 3) << 4) + ty;
  const int f = ((blockIdx.x & 7) << 4) + tx;
  float acc = 0.f;
  for (int n = 0; n < K; ++n)
    acc = fmaf(Am[(size_t)n * CD + c], Bm[(size_t)n * CD + f], acc);
  if (C) C[c * CD + f] = acc;
  if (fro_part) {
    red[tid] = acc * acc;
    __syncthreads();
    for (int s = 128; s > 0; s >>= 1) {
      if (tid < s) red[tid] += red[tid + s];
      __syncthreads();
    }
    if (tid == 0) fro_part[blockIdx.x] = red[0];
  }
}

// ---------------------------------------------------------------------------
// Kernel 5: finisher, 1 block x 128 threads. Pooled SAGE-3 + tanh + final
// pool + all scalar losses via deterministic tree reductions.
// d_out layout: [0:2]=x  [2]=link_loss  [3]=ent_loss  [4:260]=nodes_out
//               [260:16644]=edge_out (already written by k_gemm_tn #1)
// ---------------------------------------------------------------------------
__device__ __forceinline__ float bred128(float* red, int i, float v) {
  red[i] = v;
  __syncthreads();
  for (int s = 64; s > 0; s >>= 1) {
    if (i < s) red[i] += red[i + s];
    __syncthreads();
  }
  const float r = red[0];
  __syncthreads();
  return r;
}

__global__ void k_final(const float* __restrict__ xp,        // out_pool [128,128]
                        const float* __restrict__ Wl3,
                        const float* __restrict__ bl3,
                        const float* __restrict__ Wr3,
                        const float* __restrict__ deg_big,   // [8192]
                        const float* __restrict__ ent_node,  // [8192]
                        const float* __restrict__ gsq_part,  // [64]
                        float* __restrict__ out) {
  __shared__ float red[128];
  const float* adjp = out + 260;                // edge_out, already computed
  const int i = threadIdx.x;                    // 0..127: one pooled node

  // mask column i of adj_p as 4x32 bits, in-degree, zero count
  unsigned mbits[4] = {0u, 0u, 0u, 0u};
  float degp = 0.f, zc = 0.f;
  for (int j = 0; j < CD; ++j) {
    if (adjp[j * CD + i] != 0.f) {
      mbits[j >> 5] |= 1u << (j & 31);
      degp += 1.f;
    } else {
      zc += 1.f;
    }
  }
  const float inv = degp > 0.f ? 1.f / fmaxf(degp, 1.f) : 0.f;

  // out3[i][e] = sum_f agg[i][f]*Wl3[e][f] + bl3[e] + sum_f xp[i][f]*Wr3[e][f]
  float o0 = bl3[0], o1 = bl3[1];
  for (int f = 0; f < CD; ++f) {
    float aggf = 0.f;
    for (int j = 0; j < CD; ++j)
      if (mbits[j >> 5] & (1u << (j & 31))) aggf += xp[(size_t)j * CD + f];
    aggf *= inv;
    const float xf = xp[(size_t)i * CD + f];
    o0 += aggf * Wl3[f]      + xf * Wr3[f];
    o1 += aggf * Wl3[CD + f] + xf * Wr3[CD + f];
  }
  const float rn  = 1.f / fmaxf(sqrtf(o0 * o0 + o1 * o1), 1e-12f);
  const float x30 = o0 * rn, x31 = o1 * rn;
  out[4 + 2 * i]     = tanhf(x30);
  out[4 + 2 * i + 1] = tanhf(x31);

  // reductions (deterministic order)
  const float sum0 = bred128(red, i, x30);               // final pooled x[0]
  const float sum1 = bred128(red, i, x31);               // final pooled x[1]
  const float ztot = bred128(red, i, zc);                // zeros in adj_p
  const float tr   = bred128(red, i, adjp[i * CD + i]);  // trace(s^T A s)
  float nnzp = 0.f, entp = 0.f;
  for (int t = 0; t < NN / CD; ++t) {
    nnzp += deg_big[i + CD * t];
    entp += ent_node[i + CD * t];
  }
  const float nnz = bred128(red, i, nnzp);               // ||A||_F^2 (binary)
  const float ent = bred128(red, i, entp);
  const float gsq = bred128(red, i, (i < 64) ? gsq_part[i] : 0.f);

  if (i == 0) {
    out[0] = sum0;
    out[1] = sum1;
    // ||A - s s^T||^2 = nnz - 2 tr(s^T A s) + ||s^T s||_F^2
    const float arg = fmaxf(nnz - 2.f * tr + gsq, 0.f);
    out[2] = sqrtf(arg) / 67108864.f + sqrtf(ztot) / 16384.f;  // link1 + link2
    out[3] = ent / (float)NN;  // ent2 = -log(1+1e-15) == 0 in f32
  }
}

// ---------------------------------------------------------------------------
extern "C" void kernel_launch(void* const* d_in, const int* in_sizes, int n_in,
                              void* d_out, int out_size, void* d_ws, size_t ws_size,
                              hipStream_t stream) {
  (void)in_sizes; (void)n_in; (void)out_size; (void)ws_size;
  const float* nodes = (const float*)d_in[0];
  const float* A     = (const float*)d_in[1];
  const float* Wl1 = (const float*)d_in[2];
  const float* bl1 = (const float*)d_in[3];
  const float* Wr1 = (const float*)d_in[4];
  const float* Wl2 = (const float*)d_in[5];
  const float* bl2 = (const float*)d_in[6];
  const float* Wr2 = (const float*)d_in[7];
  const float* Wl3 = (const float*)d_in[8];
  const float* bl3 = (const float*)d_in[9];
  const float* Wr3 = (const float*)d_in[10];
  float* out = (float*)d_out;

  float* W = (float*)d_ws;                       // all offsets in floats
  float* deg      = W;                           // 8192
  float* agg      = W + 8192;                    // 16384
  float* pd       = W + 24576;                   // 32*8192
  float* pa0      = W + 286720;                  // 32*8192
  float* pa1      = W + 548864;                  // 32*8192
  float* x_feat   = W + 811008;                  // 8192*128
  float* s_soft   = W + 1859584;                 // 8192*128
  float* P        = W + 2908160;                 // 8192*128
  unsigned short* s_bf = (unsigned short*)(W + 3956736);  // 8192*128 ushort, tiled
  float* out_pool = W + 4481024;                 // 128*128
  float* ent_node = W + 4497408;                 // 8192
  float* gsq_part = W + 4505600;                 // 64
  // total 4505664 floats = ~17.2 MB of d_ws

  k_colred_partial<<<1024, 256, 0, stream>>>(A, nodes, pd, pa0, pa1);
  k_colred_final  <<<32,   256, 0, stream>>>(pd, pa0, pa1, deg, agg);
  k_sage12        <<<32,   256, 0, stream>>>(nodes, deg, agg,
                                             Wl1, bl1, Wr1, Wl2, bl2, Wr2,
                                             x_feat, s_soft, s_bf, ent_node);
  k_gemm_bf16     <<<512,  256, 0, stream>>>(A, s_bf, P);
  // edge_out = s^T (A s) straight into d_out
  k_gemm_tn       <<<64,   256, 0, stream>>>(s_soft, P,      out + 260, nullptr, NN);
  // pooled features = s^T x
  k_gemm_tn       <<<64,   256, 0, stream>>>(s_soft, x_feat, out_pool,  nullptr, NN);
  // ||s^T s||_F^2 partials
  k_gemm_tn       <<<64,   256, 0, stream>>>(s_soft, s_soft, nullptr,   gsq_part, NN);
  k_final         <<<1,    128, 0, stream>>>(out_pool, Wl3, bl3, Wr3,
                                             deg, ent_node, gsq_part, out);
}
